// LSTMModel_47820165874014
// MI455X (gfx1250) — matrix operand order repaired and checked by reading, not compiled, not run on hardware
//
#include <hip/hip_runtime.h>

#define TSTEPS 512
#define BATCH  512
#define HID    256

typedef __bf16 bf16;
typedef bf16  v16bf __attribute__((ext_vector_type(16)));
typedef float v8f   __attribute__((ext_vector_type(8)));
typedef int   v4i   __attribute__((ext_vector_type(4)));

union Frag { v16bf bf; v4i i2[2]; };

// ---------------------------------------------------------------------------
// Pack W [4H=1024][H=256] fp32 row-major into bf16 WMMA B-matrix tiles:
// layout [mat][ntile(64)][ktile(8)][lane(32)][elem(16)]
//   n = ntile*16 + (lane&15)
//   k = ktile*32 + (elem<8 ? (lane>=16 ? 8 : 0) + elem : 16 + (lane>=16 ? 8 : 0) + elem-8)
// ---------------------------------------------------------------------------
__global__ void pack_weights(const float* __restrict__ Whh0,
                             const float* __restrict__ Wih1,
                             const float* __restrict__ Whh1,
                             bf16* __restrict__ wsB) {
  const int perMat = 64 * 8 * 32 * 16;
  int e = blockIdx.x * blockDim.x + threadIdx.x;
  if (e >= 3 * perMat) return;
  int mat  = e / perMat;
  int r    = e % perMat;
  int nt   = r / (8 * 32 * 16);
  int kt   = (r / (32 * 16)) % 8;
  int lane = (r / 16) % 32;
  int j    = r % 16;
  int n = nt * 16 + (lane & 15);
  int k = kt * 32 + ((lane >> 4) * 8) + (j < 8 ? j : 16 + (j - 8));
  const float* W = (mat == 0) ? Whh0 : (mat == 1 ? Wih1 : Whh1);
  wsB[e] = (bf16)W[n * HID + k];
}

__device__ __forceinline__ float fsigmoid(float x) { return 1.0f / (1.0f + __expf(-x)); }
__device__ __forceinline__ float ftanh(float x)    { return 2.0f / (1.0f + __expf(-2.0f * x)) - 1.0f; }

// ---------------------------------------------------------------------------
// One pipelined GEMM pass: acc[g] += A(h) x B(weights), K=256 (8 k-steps).
// 2-stage pipeline: prefetch k-step kt+1's 4 B-fragments while WMMAs for kt
// execute; sched_barrier(0) bounds the in-flight set to one prefetch stage.
// hrow points at h[m][hi*8] (A-fragment row base for this lane).
// ---------------------------------------------------------------------------
__device__ __forceinline__ void gemm_pipe(v8f acc[4], const bf16* hrow,
                                          const bf16* wmat, const int bOff[4]) {
  Frag bcur[4], bnext[4];
#pragma unroll
  for (int g = 0; g < 4; ++g) {
    const v4i* bp = (const v4i*)(wmat + bOff[g]);
    bcur[g].i2[0] = bp[0]; bcur[g].i2[1] = bp[1];
  }
#pragma unroll
  for (int kt = 0; kt < 8; ++kt) {
    if (kt < 7) {
#pragma unroll
      for (int g = 0; g < 4; ++g) {
        const v4i* bp = (const v4i*)(wmat + bOff[g] + (kt + 1) * 512);
        bnext[g].i2[0] = bp[0]; bnext[g].i2[1] = bp[1];
      }
    }
    Frag a;
    const bf16* hp = hrow + kt * 32;
    a.i2[0] = *(const v4i*)(hp);
    a.i2[1] = *(const v4i*)(hp + 16);
#pragma unroll
    for (int g = 0; g < 4; ++g)
      acc[g] = __builtin_amdgcn_wmma_f32_16x16x32_bf16(
          false, a.bf, false, bcur[g].bf, (short)0, acc[g], false, false);
    __builtin_amdgcn_sched_barrier(0);   // hold pipeline shape, cap pressure
#pragma unroll
    for (int g = 0; g < 4; ++g) bcur[g] = bnext[g];   // SSA renames, no copies
  }
}

// ---------------------------------------------------------------------------
// Persistent 2-layer LSTM. One workgroup = 16 batch rows, 16 waves (512 thr).
// Wave w owns gate columns [16w, 16w+16) of each of the 4 gates
// (ntile g*16 + w per gate g), so i/f/g/o align per-lane for the cell update.
// A laundered integer offset keeps the weight loads loop-variant (no LICM
// hoist -> no scratch spills) while preserving GLOBAL address space.
// ---------------------------------------------------------------------------
__global__ __launch_bounds__(512, 1)
void lstm_persistent(const float* __restrict__ x,      // [B][T]
                     const float* __restrict__ wih0,   // [1024]
                     const float* __restrict__ bih0, const float* __restrict__ bhh0,
                     const float* __restrict__ bih1, const float* __restrict__ bhh1,
                     const float* __restrict__ wfc,    // [256]
                     const float* __restrict__ bfc,    // [1]
                     const bf16* __restrict__ wsB,     // packed [3][64][8][32][16]
                     float* __restrict__ out) {        // [B]
  __shared__ bf16  h0buf[2][16][HID];   // layer-0 hidden state (double buffered)
  __shared__ bf16  h1buf[2][16][HID];   // layer-1 hidden state
  __shared__ bf16  h0out[16][HID];      // layer-0 output at current t
  __shared__ float xs[2][16];           // x scalars (double buffered)
  __shared__ float lout[16];            // final fc reduction

  const int tid  = threadIdx.x;
  const int lane = tid & 31;
  const int w    = tid >> 5;       // wave 0..15
  const int m    = lane & 15;
  const int hi   = lane >> 4;
  const int b0   = blockIdx.x * 16;

  for (int idx = tid; idx < 16 * HID; idx += 512) {
    ((bf16*)h0buf[0])[idx] = (bf16)0.0f;
    ((bf16*)h1buf[0])[idx] = (bf16)0.0f;
  }
  if (tid < 16) {
    lout[tid]  = 0.0f;
    xs[0][tid] = x[(b0 + tid) * TSTEPS + 0];
  }

  // per-lane constants: n = (g*16 + w)*16 + m ; fragment base offsets per gate
  float bias0r[4], bias1r[4], w0r[4], wfcr;
  int   bOff[4];                       // element offset of (ntile, kt=0, lane)
#pragma unroll
  for (int g = 0; g < 4; ++g) {
    int nt = g * 16 + w;
    int n  = nt * 16 + m;
    bias0r[g] = bih0[n] + bhh0[n];
    bias1r[g] = bih1[n] + bhh1[n];
    w0r[g]    = wih0[n];
    bOff[g]   = nt * (8 * 32 * 16) + lane * 16;   // + kt*512 inside loop
  }
  wfcr = wfc[16 * w + m];

  float c0[8], c1[8];
#pragma unroll
  for (int v = 0; v < 8; ++v) { c0[v] = 0.0f; c1[v] = 0.0f; }

  const int perMat = 64 * 8 * 32 * 16;

  __syncthreads();

  int p = 0, xp = 0;
  for (int t = 0; t < TSTEPS; ++t) {
    // Launder an integer offset: weight addresses become loop-variant, so
    // LICM cannot hoist the weight loads across t (which previously spilled
    // them to per-wave scratch), while the GEP stays in the GLOBAL address
    // space (global_load, not flat_load).
    int toff = 0;
    asm volatile("" : "+s"(toff));
    const bf16* wsWhh0 = wsB + toff + 0 * perMat;
    const bf16* wsWih1 = wsB + toff + 1 * perMat;
    const bf16* wsWhh1 = wsB + toff + 2 * perMat;

    // ======================= layer 0 =======================
    v8f acc[4];
    float xm[8];
#pragma unroll
    for (int v = 0; v < 8; ++v) xm[v] = xs[xp][v + hi * 8];
#pragma unroll
    for (int g = 0; g < 4; ++g) {
      float base = bias0r[g], wv = w0r[g];
#pragma unroll
      for (int v = 0; v < 8; ++v) acc[g][v] = base + xm[v] * wv;
    }
    // gates0 += h0[p] @ Whh0^T
    gemm_pipe(acc, &h0buf[p][m][hi * 8], wsWhh0, bOff);

    // cell update layer 0; publish h0_new
#pragma unroll
    for (int v = 0; v < 8; ++v) {
      float ig = fsigmoid(acc[0][v]);
      float fg = fsigmoid(acc[1][v]);
      float gg = ftanh(acc[2][v]);
      float og = fsigmoid(acc[3][v]);
      float c  = fg * c0[v] + ig * gg;
      c0[v] = c;
      float h  = og * ftanh(c);
      int mm  = v + hi * 8;
      int col = 16 * w + m;
      h0buf[p ^ 1][mm][col] = (bf16)h;
      h0out[mm][col]        = (bf16)h;
    }
    if (tid < 16 && (t + 1) < TSTEPS)
      xs[xp ^ 1][tid] = x[(b0 + tid) * TSTEPS + (t + 1)];
    __syncthreads();   // h0out complete before layer-1 GEMM

    // ======================= layer 1 =======================
#pragma unroll
    for (int g = 0; g < 4; ++g) {
      float base = bias1r[g];
#pragma unroll
      for (int v = 0; v < 8; ++v) acc[g][v] = base;
    }
    // two pipelined passes keep <=1 prefetch stage (4 frags) in flight each
    gemm_pipe(acc, &h0out[m][hi * 8],     wsWih1, bOff);  // + h0out @ Wih1^T
    gemm_pipe(acc, &h1buf[p][m][hi * 8],  wsWhh1, bOff);  // + h1    @ Whh1^T

#pragma unroll
    for (int v = 0; v < 8; ++v) {
      float ig = fsigmoid(acc[0][v]);
      float fg = fsigmoid(acc[1][v]);
      float gg = ftanh(acc[2][v]);
      float og = fsigmoid(acc[3][v]);
      float c  = fg * c1[v] + ig * gg;
      c1[v] = c;
      float h  = og * ftanh(c);
      int mm  = v + hi * 8;
      int col = 16 * w + m;
      h1buf[p ^ 1][mm][col] = (bf16)h;
      if (t == TSTEPS - 1) atomicAdd(&lout[mm], h * wfcr);
    }
    __syncthreads();
    p ^= 1; xp ^= 1;
  }

  if (tid < 16) out[b0 + tid] = lout[tid] + bfc[0];
}

extern "C" void kernel_launch(void* const* d_in, const int* in_sizes, int n_in,
                              void* d_out, int out_size, void* d_ws, size_t ws_size,
                              hipStream_t stream) {
  const float* x    = (const float*)d_in[0];
  const float* Wih0 = (const float*)d_in[1];
  const float* Whh0 = (const float*)d_in[2];
  const float* bih0 = (const float*)d_in[3];
  const float* bhh0 = (const float*)d_in[4];
  const float* Wih1 = (const float*)d_in[5];
  const float* Whh1 = (const float*)d_in[6];
  const float* bih1 = (const float*)d_in[7];
  const float* bhh1 = (const float*)d_in[8];
  const float* Wfc  = (const float*)d_in[9];
  const float* bfc  = (const float*)d_in[10];

  bf16* wsB = (bf16*)d_ws;   // 3 * 512KB = 1.5MB of packed bf16 weights

  const int perMat = 64 * 8 * 32 * 16;
  int packThreads = 3 * perMat;
  pack_weights<<<(packThreads + 255) / 256, 256, 0, stream>>>(Whh0, Wih1, Whh1, wsB);

  lstm_persistent<<<BATCH / 16, 512, 0, stream>>>(
      x, Wih0, bih0, bhh0, bih1, bhh1, Wfc, bfc, wsB, (float*)d_out);
}
